// TesseraAttention_26414048870807
// MI455X (gfx1250) — compile-verified, matrix operand order checked by reading, not run on hardware
//
#include <hip/hip_runtime.h>

#define BB    2
#define TT    2048
#define HS    4096
#define HH    32
#define HHK   8
#define DD    128
#define NREP  4
#define QKV_N 6144   // HS + 2*HHK*DD
#define VOFF  5120   // HS + HHK*DD

typedef __attribute__((ext_vector_type(16))) __bf16 v16bf;
typedef __attribute__((ext_vector_type(8)))  float  v8f;

// ---------------------------------------------------------------------------
// scalar conversions
// ---------------------------------------------------------------------------
__device__ __forceinline__ unsigned short bf16u(float f) {   // f32 -> bf16 bits, RNE
  unsigned u = __builtin_bit_cast(unsigned, f);
  return (unsigned short)((u + 0x7FFFu + ((u >> 16) & 1u)) >> 16);
}
__device__ __forceinline__ __bf16 bf16_of(float f) {
  unsigned short s = bf16u(f);
  return __builtin_bit_cast(__bf16, s);
}
__device__ __forceinline__ float bf2f(__bf16 b) {
  unsigned u = (unsigned)__builtin_bit_cast(unsigned short, b) << 16;
  return __builtin_bit_cast(float, u);
}

// ---------------------------------------------------------------------------
// 16x32 bf16 fragment, CDNA5 A-matrix layout, source bf16 row-major with the
// contraction dim contiguous. lane L: row = row0 + (L&15); half = L>>4;
// v[0..3] <- K = half*8 + {0..7}; v[4..7] <- K = 16 + half*8 + {0..7}.
// Two 16B loads per lane. Used for B too when B is stored [N,K] row-major.
// ---------------------------------------------------------------------------
__device__ __forceinline__ v16bf load_frag_bf(const __bf16* __restrict__ base,
                                              int ld, int row0, int k0, int lane) {
  const __bf16* p = base + (size_t)(row0 + (lane & 15)) * ld + k0 + ((lane >> 4) << 3);
  union { v16bf v; uint4 q[2]; } u;
  u.q[0] = *(const uint4*)p;         // elems 0..7
  u.q[1] = *(const uint4*)(p + 16);  // elems 8..15
  return u.v;
}

// ---------------------------------------------------------------------------
// Streaming f32 -> bf16 conversion, 8 elems/thread, packed 16B stores.
// ---------------------------------------------------------------------------
__global__ void __launch_bounds__(256)
cvt_f32_bf16(const float* __restrict__ in, unsigned short* __restrict__ out, size_t n) {
  size_t i = ((size_t)blockIdx.x * 256 + threadIdx.x) * 8;
  if (i >= n) return;
  float4 a = *(const float4*)(in + i);
  float4 b = *(const float4*)(in + i + 4);
  uint4 o;
  o.x = (unsigned)bf16u(a.x) | ((unsigned)bf16u(a.y) << 16);
  o.y = (unsigned)bf16u(a.z) | ((unsigned)bf16u(a.w) << 16);
  o.z = (unsigned)bf16u(b.x) | ((unsigned)bf16u(b.y) << 16);
  o.w = (unsigned)bf16u(b.z) | ((unsigned)bf16u(b.w) << 16);
  *(uint4*)(out + i) = o;
}

// ---------------------------------------------------------------------------
// GEMM: C[M,N] = A[M,K] @ B[N,K]^T, bf16 inputs, f32 accumulate.
// Block = 256 threads = 8 waves; block tile 128x128; wave tile 32x64.
// ---------------------------------------------------------------------------
template <bool STORE_BF16>
__global__ void __launch_bounds__(256)
gemm_bf16(const unsigned short* __restrict__ Ap, const unsigned short* __restrict__ Bp,
          void* __restrict__ Cv, int M, int N, int K) {
  const __bf16* A = (const __bf16*)Ap;
  const __bf16* B = (const __bf16*)Bp;
  const int lane = threadIdx.x & 31;
  const int wave = threadIdx.x >> 5;   // 0..7
  const int wm = wave >> 1;            // 0..3  (M direction)
  const int wn = wave & 1;             // 0..1  (N direction)
  const int m0 = blockIdx.x * 128 + wm * 32;
  const int n0 = blockIdx.y * 128 + wn * 64;

  v8f z = {};
  v8f acc[2][4];
#pragma unroll
  for (int a = 0; a < 2; ++a)
#pragma unroll
    for (int b = 0; b < 4; ++b) acc[a][b] = z;

  for (int k0 = 0; k0 < K; k0 += 32) {
    if (k0 + 32 < K) {  // L2 prefetch of next K-slab -> global_prefetch_b8
      __builtin_prefetch(A + (size_t)(m0 + (lane & 15)) * K + k0 + 32, 0, 1);
      __builtin_prefetch(B + (size_t)(n0 + (lane & 15)) * K + k0 + 32, 0, 1);
    }
    v16bf af[2], bfr[4];
#pragma unroll
    for (int a = 0; a < 2; ++a) af[a] = load_frag_bf(A, K, m0 + a * 16, k0, lane);
#pragma unroll
    for (int b = 0; b < 4; ++b) bfr[b] = load_frag_bf(B, K, n0 + b * 16, k0, lane);
#pragma unroll
    for (int a = 0; a < 2; ++a)
#pragma unroll
      for (int b = 0; b < 4; ++b)
        acc[a][b] = __builtin_amdgcn_wmma_f32_16x16x32_bf16(
            false, af[a], false, bfr[b], (short)0, acc[a][b], false, false);
  }

  const int cn = lane & 15, half = lane >> 4;
#pragma unroll
  for (int a = 0; a < 2; ++a)
#pragma unroll
    for (int i = 0; i < 8; ++i) {
      int row = m0 + a * 16 + i + 8 * half;   // C layout: VGPR i -> row i + 8*half
      if (STORE_BF16) {
        __bf16* cp = (__bf16*)Cv + (size_t)row * N + n0 + cn;
#pragma unroll
        for (int b = 0; b < 4; ++b) cp[b * 16] = bf16_of(acc[a][b][i]);
      } else {
        float* cp = (float*)Cv + (size_t)row * N + n0 + cn;
#pragma unroll
        for (int b = 0; b < 4; ++b) cp[b * 16] = acc[a][b][i];
      }
    }
}

// ---------------------------------------------------------------------------
// RoPE (rotate-half) in place on the bf16 qkv buffer; q additionally scaled
// by 1/sqrt(D) so flash attention needs no per-score multiply.
// ---------------------------------------------------------------------------
__global__ void __launch_bounds__(256)
rope_bf16(unsigned short* __restrict__ qkvp, const float* __restrict__ cosb,
          const float* __restrict__ sinb, float qscale) {
  __bf16* qkv = (__bf16*)qkvp;
  size_t idx = (size_t)blockIdx.x * blockDim.x + threadIdx.x;
  const size_t total = (size_t)BB * TT * (HH + HHK) * (DD / 2);
  if (idx >= total) return;
  int d = (int)(idx & 63);
  size_t r = idx >> 6;
  int hh = (int)(r % (HH + HHK)); r /= (HH + HHK);
  int t  = (int)(r % TT);
  int b  = (int)(r / TT);
  int off = (hh < HH) ? hh * DD : HS + (hh - HH) * DD;
  float sc = (hh < HH) ? qscale : 1.0f;
  __bf16* p = qkv + ((size_t)b * TT + t) * QKV_N + off;
  float c1 = cosb[t * DD + d],      s1 = sinb[t * DD + d];
  float c2 = cosb[t * DD + d + 64], s2 = sinb[t * DD + d + 64];
  float u1 = bf2f(p[d]), u2 = bf2f(p[d + 64]);
  p[d]      = bf16_of((u1 * c1 - u2 * s1) * sc);
  p[d + 64] = bf16_of((u2 * c2 + u1 * s2) * sc);
}

// ---------------------------------------------------------------------------
// V transpose: qkv v-slice [b][t][kh*D+d] -> vT[b][kh][d][t]  (bf16, 8 MB).
// LDS-tiled 32x32; coalesced on both sides.
// ---------------------------------------------------------------------------
__global__ void __launch_bounds__(256)
transpose_v(const unsigned short* __restrict__ qkvp, unsigned short* __restrict__ vTp) {
  __shared__ unsigned short tile[32][33];
  const int bh = blockIdx.z;                 // b*HHK + kh
  const int t0 = blockIdx.x * 32;
  const int d0 = blockIdx.y * 32;
  const int tx = threadIdx.x & 31;
  const int ty = threadIdx.x >> 5;           // 0..7
  const int b = bh / HHK, kh = bh % HHK;
  const unsigned short* src = qkvp + (size_t)b * TT * QKV_N + VOFF + kh * DD;
#pragma unroll
  for (int j = 0; j < 4; ++j) {
    int tl = ty + 8 * j;
    tile[tl][tx] = src[(size_t)(t0 + tl) * QKV_N + d0 + tx];
  }
  __syncthreads();
  unsigned short* dst = vTp + (size_t)bh * DD * TT;
#pragma unroll
  for (int j = 0; j < 4; ++j) {
    int dl = ty + 8 * j;
    dst[(size_t)(d0 + dl) * TT + t0 + tx] = tile[tx][dl];
  }
}

// ---------------------------------------------------------------------------
// Causal GQA flash attention, bf16 in / bf16 out. One wave32 per
// (b, h, 16-query tile). Q resident as 4 A-fragments; 32-key chunks; online
// softmax in f32; P staged through 1 KB LDS into A-layout for P@V.
// Mask only applied on the diagonal chunk (wave-uniform branch).
// ---------------------------------------------------------------------------
__global__ void __launch_bounds__(32)
flash_attn(const unsigned short* __restrict__ qkvp, const unsigned short* __restrict__ vTp,
           unsigned short* __restrict__ attp) {
  const __bf16* qkv = (const __bf16*)qkvp;
  __bf16* att = (__bf16*)attp;
  __shared__ __bf16 pbuf[16 * 32];
  const int lane = threadIdx.x;
  const int nqt = TT / 16;
  const int qt = blockIdx.x % nqt;
  const int h  = (blockIdx.x / nqt) % HH;
  const int b  = blockIdx.x / (nqt * HH);
  const int kh = h / NREP;
  const int q0 = qt * 16;
  const __bf16* qb = qkv + (size_t)b * TT * QKV_N + h * DD;
  const __bf16* kb = qkv + (size_t)b * TT * QKV_N + HS + kh * DD;
  const __bf16* vb = (const __bf16*)vTp + (size_t)(b * HHK + kh) * DD * TT; // [d][t]

  v16bf qf[4];
#pragma unroll
  for (int c = 0; c < 4; ++c) qf[c] = load_frag_bf(qb, QKV_N, q0, c * 32, lane);

  v8f z = {};
  v8f o[8];
#pragma unroll
  for (int d = 0; d < 8; ++d) o[d] = z;
  float rm[8], rl[8];
#pragma unroll
  for (int i = 0; i < 8; ++i) { rm[i] = -3.0e38f; rl[i] = 0.f; }

  const int half = lane >> 4, cn = lane & 15;
  const int nkc = (q0 + 16 + 31) / 32;       // causal: keys < q0+16

  for (int kc = 0; kc < nkc; ++kc) {
    const int key0 = kc * 32;

    // ---- load all K fragments (16 b128 loads), then 8 WMMAs ----
    v16bf kf[2][4];
#pragma unroll
    for (int t = 0; t < 2; ++t)
#pragma unroll
      for (int c = 0; c < 4; ++c)
        kf[t][c] = load_frag_bf(kb, QKV_N, key0 + t * 16, c * 32, lane);

    v8f s[2]; s[0] = z; s[1] = z;
#pragma unroll
    for (int t = 0; t < 2; ++t)
#pragma unroll
      for (int c = 0; c < 4; ++c)
        s[t] = __builtin_amdgcn_wmma_f32_16x16x32_bf16(
            false, qf[c], false, kf[t][c], (short)0, s[t], false, false);

    // ---- V fragments issued early (b128 from vT) to overlap softmax ----
    v16bf vf[8];
#pragma unroll
    for (int d = 0; d < 8; ++d)
      vf[d] = load_frag_bf(vb, TT, d * 16, key0, lane);

    // ---- causal mask: only the diagonal chunk needs it (uniform branch) ----
    if (key0 + 31 > q0) {
#pragma unroll
      for (int i = 0; i < 8; ++i) {
        const int row = q0 + i + 8 * half;
#pragma unroll
        for (int t = 0; t < 2; ++t) {
          int key = key0 + t * 16 + cn;
          if (key > row) s[t][i] = -3.0e38f;
        }
      }
    }

    // ---- online softmax update (f32) ----
    float nm[8], alpha[8];
#pragma unroll
    for (int i = 0; i < 8; ++i) {
      float mx = fmaxf(s[0][i], s[1][i]);
#pragma unroll
      for (int d = 1; d < 16; d <<= 1) mx = fmaxf(mx, __shfl_xor(mx, d, 16));
      nm[i] = fmaxf(rm[i], mx);
    }
#pragma unroll
    for (int i = 0; i < 8; ++i) {
      alpha[i] = __expf(rm[i] - nm[i]);
      float ps = 0.f;
#pragma unroll
      for (int t = 0; t < 2; ++t) {
        float p = __expf(s[t][i] - nm[i]);
        s[t][i] = p;
        ps += p;
      }
#pragma unroll
      for (int d = 1; d < 16; d <<= 1) ps += __shfl_xor(ps, d, 16);
      rl[i] = rl[i] * alpha[i] + ps;
      rm[i] = nm[i];
    }
#pragma unroll
    for (int d = 0; d < 8; ++d)
#pragma unroll
      for (int i = 0; i < 8; ++i) o[d][i] *= alpha[i];

    // ---- C-layout P -> LDS -> A-layout fragment (2x b128 LDS reads) ----
    __syncthreads();
#pragma unroll
    for (int t = 0; t < 2; ++t)
#pragma unroll
      for (int i = 0; i < 8; ++i)
        pbuf[(i + 8 * half) * 32 + t * 16 + cn] = bf16_of(s[t][i]);
    __syncthreads();
    v16bf pf = load_frag_bf(pbuf, 32, 0, 0, lane);

    // ---- O += P @ V : 8 WMMAs over D-subtiles ----
#pragma unroll
    for (int d = 0; d < 8; ++d)
      o[d] = __builtin_amdgcn_wmma_f32_16x16x32_bf16(
          false, pf, false, vf[d], (short)0, o[d], false, false);
  }

  // epilogue: att[b, q0+row, h, :] = O / l   (bf16)
#pragma unroll
  for (int i = 0; i < 8; ++i) {
    const int row = q0 + i + 8 * half;
    const float inv = 1.0f / rl[i];
    __bf16* op = att + (((size_t)b * TT + row) * HH + h) * DD + cn;
#pragma unroll
    for (int d = 0; d < 8; ++d) op[d * 16] = bf16_of(o[d][i] * inv);
  }
}

// ---------------------------------------------------------------------------
// Launch: 3x cvt -> qkv-GEMM(bf16) -> RoPE -> V-transpose -> flash -> out-GEMM
// ---------------------------------------------------------------------------
extern "C" void kernel_launch(void* const* d_in, const int* in_sizes, int n_in,
                              void* d_out, int out_size, void* d_ws, size_t ws_size,
                              hipStream_t stream) {
  const float* x     = (const float*)d_in[0];
  const float* rc    = (const float*)d_in[1];
  const float* rs    = (const float*)d_in[2];
  const float* Wqkv  = (const float*)d_in[3];
  const float* Wproj = (const float*)d_in[4];

  const int M = BB * TT;  // 4096
  const size_t xN  = (size_t)M * HS;             // 16M
  const size_t wqN = (size_t)QKV_N * HS;         // 24M
  const size_t wpN = (size_t)HS * HS;            // 16M
  const size_t qkN = (size_t)M * QKV_N;          // 24M
  const size_t atN = (size_t)M * HS;             // 16M

  unsigned short* xb   = (unsigned short*)d_ws;
  unsigned short* wqb  = xb + xN;
  unsigned short* wpb  = wqb + wqN;
  unsigned short* qkvb = wpb + wpN;
  unsigned short* attb = qkvb + qkN;
  unsigned short* vT   = attb + atN;             // total 100M bf16 = 200 MB

  cvt_f32_bf16<<<dim3((unsigned)(xN / 8 / 256)),  dim3(256), 0, stream>>>(x, xb, xN);
  cvt_f32_bf16<<<dim3((unsigned)(wqN / 8 / 256)), dim3(256), 0, stream>>>(Wqkv, wqb, wqN);
  cvt_f32_bf16<<<dim3((unsigned)(wpN / 8 / 256)), dim3(256), 0, stream>>>(Wproj, wpb, wpN);

  gemm_bf16<true><<<dim3(M / 128, QKV_N / 128), dim3(256), 0, stream>>>(
      xb, wqb, qkvb, M, QKV_N, HS);

  const size_t rt = (size_t)BB * TT * (HH + HHK) * (DD / 2);
  rope_bf16<<<dim3((unsigned)((rt + 255) / 256)), dim3(256), 0, stream>>>(
      qkvb, rc, rs, 0.08838834764831845f /* 1/sqrt(128) */);

  transpose_v<<<dim3(TT / 32, DD / 32, BB * HHK), dim3(256), 0, stream>>>(qkvb, vT);

  flash_attn<<<dim3(BB * HH * (TT / 16)), dim3(32), 0, stream>>>(qkvb, vT, attb);

  gemm_bf16<false><<<dim3(M / 128, HS / 128), dim3(256), 0, stream>>>(
      attb, wpb, d_out, M, HS, HS);
}
